// APPNP_30562987278368
// MI455X (gfx1250) — matrix operand order repaired and checked by reading, not compiled, not run on hardware
//
#include <hip/hip_runtime.h>
#include <hip/hip_bf16.h>

// ---------------------------------------------------------------------------
// APPNP: 3-layer MLP (bf16 WMMA GEMMs, f32 accumulate) + K=10 propagation
// rounds. Propagation uses a CSR-by-destination built once per launch so the
// per-iteration kernel is pure gathers (L2-resident) with zero atomics.
// ---------------------------------------------------------------------------

#define IN_C   500
#define IN_CP  512          // padded K for layer 1 (multiple of 32)
#define HID_C  256
#define OUT_C  48
#define K_ITER 10
#define ALPHA_F 0.1f
#define SCAN_B 256

typedef __bf16 bf16_t;
typedef bf16_t v16bf __attribute__((ext_vector_type(16)));
typedef bf16_t v8bf  __attribute__((ext_vector_type(8)));
typedef float  v8f   __attribute__((ext_vector_type(8)));

// A-matrix fragment (16x32 MxK, bf16): lane<16 holds K = k0+{0..15},
// lane>=16 holds K = k0+16+{0..15}   (same k order as B fragment)
__device__ __forceinline__ v16bf load_a_frag(const bf16_t* __restrict__ p, int half) {
  return *(const v16bf*)(p + half * 16); // 32B -> 2x global_load_b128
}

// B-matrix fragment (32x16 KxN, bf16): lane holds one column of B = one row
// of W (C = A @ W^T). lane<16: K = k0+{0..15}; lane>=16: K = k0+16+{0..15}.
__device__ __forceinline__ v16bf load_b_frag(const bf16_t* __restrict__ p, int half) {
  return *(const v16bf*)(p + half * 16); // 32B -> 2x global_load_b128
}

// One wave computes a 16 x (16*NTW) strip of C = act(A @ W^T + b).
// A loaded once per k-step, feeding NTW independent WMMA chains.
// A: [M,K] bf16 row-major, W: [N,K] bf16 row-major. K % 32 == 0.
template <int NTW, bool OBF16>
__global__ void gemm_bf16_wmma(const bf16_t* __restrict__ A,
                               const bf16_t* __restrict__ W,
                               const float* __restrict__ bias,
                               void* __restrict__ Cout,
                               int N, int K, int relu,
                               int ngrp, int total_waves) {
  const int wave = blockIdx.x * (blockDim.x >> 5) + (threadIdx.x >> 5);
  if (wave >= total_waves) return;          // wave-uniform: EXEC stays all-1s
  const int lane = threadIdx.x & 31;
  const int mt = wave / ngrp;
  const int g  = wave % ngrp;
  const int m0 = mt << 4;
  const int n0 = g * (16 * NTW);
  const int half = lane >> 4, l = lane & 15;

  const bf16_t* __restrict__ arow = A + (size_t)(m0 + l) * K;
  const bf16_t* wrow[NTW];
#pragma unroll
  for (int t = 0; t < NTW; ++t)
    wrow[t] = W + (size_t)(n0 + t * 16 + l) * K;

  v8f acc[NTW];
#pragma unroll
  for (int t = 0; t < NTW; ++t) acc[t] = {};

  for (int k0 = 0; k0 < K; k0 += 32) {
    __builtin_prefetch(arow + k0 + 128, 0, 1);   // global_prefetch_b8
    v16bf a = load_a_frag(arow + k0, half);
    v16bf b[NTW];
#pragma unroll
    for (int t = 0; t < NTW; ++t) b[t] = load_b_frag(wrow[t] + k0, half);
#pragma unroll
    for (int t = 0; t < NTW; ++t)
      acc[t] = __builtin_amdgcn_wmma_f32_16x16x32_bf16(
          false, a, false, b[t], (short)0, acc[t], false, false);
  }

  const int r0 = m0 + half * 8;               // C layout: VGPR j -> row r0+j
#pragma unroll
  for (int t = 0; t < NTW; ++t) {
    const int col = n0 + t * 16 + l;
    const float bv = bias[col];
#pragma unroll
    for (int j = 0; j < 8; ++j) {
      float v = acc[t][j] + bv;
      if (relu) v = fmaxf(v, 0.0f);
      if (OBF16)
        ((bf16_t*)Cout)[(size_t)(r0 + j) * N + col] = (bf16_t)v;
      else
        ((float*)Cout)[(size_t)(r0 + j) * N + col] = v;
    }
  }
}

// f32 -> bf16 with row padding: out[r*Kout+k] = (k<Kin) ? in[r*Kin+k] : 0
__global__ void k_cvt_pad(const float* __restrict__ in, bf16_t* __restrict__ out,
                          int rows, int Kin, int Kout) {
  int i = blockIdx.x * blockDim.x + threadIdx.x;
  if (i >= rows * Kout) return;
  int r = i / Kout, k = i - r * Kout;
  out[i] = (k < Kin) ? (bf16_t)in[(size_t)r * Kin + k] : (bf16_t)0.0f;
}

// ------------------------- CSR construction --------------------------------

__global__ void k_zero_i32(int* __restrict__ p, int n) {
  int i = blockIdx.x * blockDim.x + threadIdx.x;
  if (i < n) p[i] = 0;
}

__global__ void k_count_accum(const int* __restrict__ dst, int* __restrict__ cnt, int E) {
  int e = blockIdx.x * blockDim.x + threadIdx.x;
  if (e < E) atomicAdd(&cnt[dst[e]], 1);
}

// dinv[i] = rsqrt(in_degree + 1 self loop)
__global__ void k_dinv(const int* __restrict__ cnt, float* __restrict__ dinv, int n) {
  int i = blockIdx.x * blockDim.x + threadIdx.x;
  if (i < n) dinv[i] = rsqrtf((float)cnt[i] + 1.0f);
}

// scan pass 1: per-block sums of counts
__global__ void k_scan_block(const int* __restrict__ cnt, int* __restrict__ bsum, int n) {
  __shared__ int sm[SCAN_B];
  int i = blockIdx.x * SCAN_B + threadIdx.x;
  sm[threadIdx.x] = (i < n) ? cnt[i] : 0;
  __syncthreads();
  for (int s = SCAN_B / 2; s > 0; s >>= 1) {
    if (threadIdx.x < s) sm[threadIdx.x] += sm[threadIdx.x + s];
    __syncthreads();
  }
  if (threadIdx.x == 0) bsum[blockIdx.x] = sm[0];
}

// scan pass 2: single-block exclusive scan of block sums (nb <= SCAN_B)
__global__ void k_scan_sums(int* __restrict__ bsum, int nb) {
  __shared__ int sm[SCAN_B];
  int v = (threadIdx.x < nb) ? bsum[threadIdx.x] : 0;
  sm[threadIdx.x] = v;
  __syncthreads();
  for (int off = 1; off < SCAN_B; off <<= 1) {
    int t = (threadIdx.x >= off) ? sm[threadIdx.x - off] : 0;
    __syncthreads();
    sm[threadIdx.x] += t;
    __syncthreads();
  }
  if (threadIdx.x < nb) bsum[threadIdx.x] = sm[threadIdx.x] - v;  // exclusive
}

// scan pass 3: per-block exclusive scan + block offset -> rowptr[0..n]
__global__ void k_scan_final(const int* __restrict__ cnt, const int* __restrict__ bsum,
                             int* __restrict__ rowptr, int n) {
  __shared__ int sm[SCAN_B];
  int i = blockIdx.x * SCAN_B + threadIdx.x;
  int v = (i < n) ? cnt[i] : 0;
  sm[threadIdx.x] = v;
  __syncthreads();
  for (int off = 1; off < SCAN_B; off <<= 1) {
    int t = (threadIdx.x >= off) ? sm[threadIdx.x - off] : 0;
    __syncthreads();
    sm[threadIdx.x] += t;
    __syncthreads();
  }
  int incl = sm[threadIdx.x];
  if (i < n) rowptr[i] = bsum[blockIdx.x] + incl - v;      // exclusive
  if (i == n - 1) rowptr[n] = bsum[blockIdx.x] + incl;     // total = E
}

// slot edges into per-destination lists; precompute edge norm
__global__ void k_csr_fill(const int* __restrict__ src, const int* __restrict__ dst,
                           const int* __restrict__ rowptr, int* __restrict__ cursor,
                           const float* __restrict__ dinv,
                           int* __restrict__ csr_src, float* __restrict__ csr_w, int E) {
  int e = blockIdx.x * blockDim.x + threadIdx.x;
  if (e >= E) return;
  int d = dst[e], s = src[e];
  int pos = rowptr[d] + atomicAdd(&cursor[d], 1);
  csr_src[pos] = s;
  csr_w[pos] = dinv[s] * dinv[d];
}

// ------------------------- fused APPNP iteration ----------------------------
// 16 lanes per node, 3 contiguous components per lane. Pure gathers from the
// L2-resident hcur; register accumulation; self-loop + alpha-mix folded in.
__global__ void k_appnp_iter(const int* __restrict__ rowptr,
                             const int* __restrict__ csr_src,
                             const float* __restrict__ csr_w,
                             const float* __restrict__ dinv,
                             const float* __restrict__ hcur,
                             const float* __restrict__ h0,
                             float* __restrict__ hnext, int Nn) {
  int t = blockIdx.x * blockDim.x + threadIdx.x;
  int node = t >> 4;
  if (node >= Nn) return;
  int part = (t & 15) * 3;
  int e0 = rowptr[node], e1 = rowptr[node + 1];
  float a0 = 0.0f, a1 = 0.0f, a2 = 0.0f;
  for (int e = e0; e < e1; ++e) {
    int s = csr_src[e];
    float w = csr_w[e];
    const float* hp = hcur + (size_t)s * OUT_C + part;
    a0 += w * hp[0];
    a1 += w * hp[1];
    a2 += w * hp[2];
  }
  float di = dinv[node], dd = di * di;
  const float* hc = hcur + (size_t)node * OUT_C + part;
  const float* hz = h0 + (size_t)node * OUT_C + part;
  float* o = hnext + (size_t)node * OUT_C + part;
  o[0] = (1.0f - ALPHA_F) * (a0 + hc[0] * dd) + ALPHA_F * hz[0];
  o[1] = (1.0f - ALPHA_F) * (a1 + hc[1] * dd) + ALPHA_F * hz[1];
  o[2] = (1.0f - ALPHA_F) * (a2 + hc[2] * dd) + ALPHA_F * hz[2];
}

// ---------------------------------------------------------------------------

extern "C" void kernel_launch(void* const* d_in, const int* in_sizes, int n_in,
                              void* d_out, int out_size, void* d_ws, size_t ws_size,
                              hipStream_t stream) {
  const float* x  = (const float*)d_in[0];
  const int*   ei = (const int*)d_in[1];
  const float* W1 = (const float*)d_in[2];
  const float* b1 = (const float*)d_in[3];
  const float* W2 = (const float*)d_in[4];
  const float* b2 = (const float*)d_in[5];
  const float* W3 = (const float*)d_in[6];
  const float* b3 = (const float*)d_in[7];

  const int Nn = in_sizes[0] / IN_C;   // 50000 (divisible by 16)
  const int E  = in_sizes[1] / 2;      // 1.6M
  const int* src = ei;
  const int* dst = ei + E;

  auto al = [](size_t v) { return (v + 255) & ~(size_t)255; };

  // ---- workspace layout (dead-buffer reuse, peak ~77.3 MB) ----
  char* ws = (char*)d_ws;
  const size_t szXB = al((size_t)Nn * IN_CP * sizeof(bf16_t));     // 51.2 MB
  const size_t szW1 = al((size_t)HID_C * IN_CP * sizeof(bf16_t));
  const size_t szW2 = al((size_t)HID_C * HID_C * sizeof(bf16_t));
  const size_t szW3 = al((size_t)OUT_C * HID_C * sizeof(bf16_t));
  const size_t szO  = al((size_t)Nn * OUT_C * sizeof(float));      //  9.6 MB
  const size_t szNI = al((size_t)(Nn + 1) * sizeof(int));          //  0.2 MB
  const size_t szEI = al((size_t)E * sizeof(int));                 //  6.4 MB

  bf16_t* xb  = (bf16_t*)(ws);                                     // [0, 51.2M)
  bf16_t* w1b = (bf16_t*)(ws + szXB);
  bf16_t* w2b = (bf16_t*)(ws + szXB + szW1);
  bf16_t* w3b = (bf16_t*)(ws + szXB + szW1 + szW2);
  bf16_t* h1b = (bf16_t*)(ws + szXB + szW1 + szW2 + szW3);         // +25.6M
  bf16_t* h2b = (bf16_t*)(ws);                                     // over dead xb
  float*  h0  = (float*)(h1b);                                     // over dead h1b

  // propagation-phase buffers live in the dead xb/h2b region (51.2 MB)
  char* pr = ws;
  float* bufA   = (float*)(pr);                pr += szO;
  float* bufB   = (float*)(pr);                pr += szO;
  float* dinv   = (float*)(pr);                pr += szNI;
  int*   cnt    = (int*)(pr);                  pr += szNI;   // counts, then cursor
  int*   rowptr = (int*)(pr);                  pr += szNI;
  int*   bsum   = (int*)(pr);                  pr += al(SCAN_B * sizeof(int));
  int*   csr_s  = (int*)(pr);                  pr += szEI;
  float* csr_w  = (float*)(pr);                pr += szEI;   // ends ~33 MB

  const int TPB = 256;                      // 8 waves/block

  // ---- bf16 conversion / K padding (one-shot, streaming) ----
  { int n = Nn * IN_CP;
    k_cvt_pad<<<(n + TPB - 1) / TPB, TPB, 0, stream>>>(x, xb, Nn, IN_C, IN_CP); }
  { int n = HID_C * IN_CP;
    k_cvt_pad<<<(n + TPB - 1) / TPB, TPB, 0, stream>>>(W1, w1b, HID_C, IN_C, IN_CP); }
  { int n = HID_C * HID_C;
    k_cvt_pad<<<(n + TPB - 1) / TPB, TPB, 0, stream>>>(W2, w2b, HID_C, HID_C, HID_C); }
  { int n = OUT_C * HID_C;
    k_cvt_pad<<<(n + TPB - 1) / TPB, TPB, 0, stream>>>(W3, w3b, OUT_C, HID_C, HID_C); }

  // ---- MLP encoder: three WMMA GEMMs ----
  { int mt = Nn / 16, ngrp = HID_C / (16 * 4), waves = mt * ngrp;   // NTW=4
    gemm_bf16_wmma<4, true><<<(waves + 7) / 8, TPB, 0, stream>>>(
        xb, w1b, b1, h1b, HID_C, IN_CP, 1, ngrp, waves); }
  { int mt = Nn / 16, ngrp = HID_C / (16 * 4), waves = mt * ngrp;   // NTW=4
    gemm_bf16_wmma<4, true><<<(waves + 7) / 8, TPB, 0, stream>>>(
        h1b, w2b, b2, h2b, HID_C, HID_C, 1, ngrp, waves); }
  { int mt = Nn / 16, ngrp = 1, waves = mt;                         // NTW=3
    gemm_bf16_wmma<3, false><<<(waves + 7) / 8, TPB, 0, stream>>>(
        h2b, w3b, b3, h0, OUT_C, HID_C, 0, ngrp, waves); }

  // ---- CSR-by-destination build (one-time; xb/h2b region now dead) ----
  const int nblk = (Nn + SCAN_B - 1) / SCAN_B;                      // 196
  k_zero_i32<<<(Nn + TPB - 1) / TPB, TPB, 0, stream>>>(cnt, Nn);
  k_count_accum<<<(E + TPB - 1) / TPB, TPB, 0, stream>>>(dst, cnt, E);
  k_dinv<<<(Nn + TPB - 1) / TPB, TPB, 0, stream>>>(cnt, dinv, Nn);
  k_scan_block<<<nblk, SCAN_B, 0, stream>>>(cnt, bsum, Nn);
  k_scan_sums<<<1, SCAN_B, 0, stream>>>(bsum, nblk);
  k_scan_final<<<nblk, SCAN_B, 0, stream>>>(cnt, bsum, rowptr, Nn);
  k_zero_i32<<<(Nn + TPB - 1) / TPB, TPB, 0, stream>>>(cnt, Nn);   // -> cursor
  k_csr_fill<<<(E + TPB - 1) / TPB, TPB, 0, stream>>>(src, dst, rowptr, cnt,
                                                      dinv, csr_s, csr_w, E);

  // ---- K propagation rounds: atomic-free gathers over L2-resident state ----
  const float* hcur = h0;
  const int nthr = Nn * 16;
  for (int it = 0; it < K_ITER; ++it) {
    float* hnext = (it == K_ITER - 1) ? (float*)d_out
                                      : ((it & 1) ? bufB : bufA);
    k_appnp_iter<<<(nthr + TPB - 1) / TPB, TPB, 0, stream>>>(
        rowptr, csr_s, csr_w, dinv, hcur, h0, hnext, Nn);
    hcur = hnext;
  }
  (void)n_in; (void)out_size; (void)ws_size;
}
